// GraphModel_69337952026806
// MI455X (gfx1250) — compile-verified, matrix operand order checked by reading, not run on hardware
//
#include <hip/hip_runtime.h>

// ---------------------------------------------------------------------------
// Types for CDNA5 WMMA (gfx1250, wave32)
// ---------------------------------------------------------------------------
typedef __attribute__((ext_vector_type(16))) __bf16 v16bf;
typedef __attribute__((ext_vector_type(8)))  __bf16 v8bf;
typedef __attribute__((ext_vector_type(8)))  float  v8f;

#define N_CLS   12
#define N_EDGE  144           // 12*12
#define C_DIM   512
#define BATCH   2048
#define X_ROWS  (BATCH * N_CLS)     // 24576
#define E_ROWS  (BATCH * N_EDGE)    // 294912
#define BN_EPS  1e-5f

static __device__ __forceinline__ unsigned short f2bf(float f) {
    union { float f; unsigned int u; } c; c.f = f;
    unsigned int u = c.u;
    unsigned int r = u + 0x7FFFu + ((u >> 16) & 1u);   // round-to-nearest-even
    return (unsigned short)(r >> 16);
}

// ---------------------------------------------------------------------------
// fp32 -> bf16 bulk convert (vectorized x4)
// ---------------------------------------------------------------------------
__global__ void cvt_f32_bf16_v4(const float4* __restrict__ in,
                                ushort4* __restrict__ out, size_t n4) {
    size_t i = (size_t)blockIdx.x * blockDim.x + threadIdx.x;
    size_t stride = (size_t)gridDim.x * blockDim.x;
    for (; i < n4; i += stride) {
        float4 v = in[i];
        ushort4 o;
        o.x = f2bf(v.x); o.y = f2bf(v.y); o.z = f2bf(v.z); o.w = f2bf(v.w);
        out[i] = o;
    }
}

// ---------------------------------------------------------------------------
// WMMA GEMM: out[M,512] = A[M,512](bf16) @ W[512,512]^T (bf16), fp32 out.
// Block = 128 threads (4 waves). Block tile 128(M) x 64(N); wave tile 32 x 64:
// two A fragments (rows m..m+15, m+16..m+31) reuse each B fragment, giving
// 8 WMMAs per k-step against 8 ds_load_b128 + 4 global_load_b128.
// W tile (64 output cols x 512 K) staged in LDS (64 KB), shared by all waves.
// Optional fused epilogue: out += fA[b,i,c] + fB[b,j,c] with m = b*144+i*12+j.
// ---------------------------------------------------------------------------
__global__ __launch_bounds__(128)
void gemm_bf16_wmma(const __bf16* __restrict__ A, const __bf16* __restrict__ W,
                    float* __restrict__ out, int Mrows,
                    const float* __restrict__ fA, const float* __restrict__ fB,
                    int fuse) {
    __shared__ __bf16 sW[64 * C_DIM];                 // 65536 bytes
    const int tid = threadIdx.x;
    const int n0  = blockIdx.y * 64;

    // Cooperative load: 64 contiguous rows of W (= 64 KB) into LDS.
    {
        const uint4* gw = (const uint4*)(W + (size_t)n0 * C_DIM);
        uint4* lw = (uint4*)sW;
        #pragma unroll
        for (int i = 0; i < 32; ++i)                  // 4096 x 16B chunks / 128 thr
            lw[tid + i * 128] = gw[tid + i * 128];
    }
    __syncthreads();

    const int lane = tid & 31;
    const int wave = tid >> 5;
    const int m0   = blockIdx.x * 128 + wave * 32;
    const int row  = lane & 15;
    const int hi   = lane >> 4;                       // 0 or 1
    const int sel  = hi * 8;                          // K sub-select per ISA layout

    const __bf16* aRow0 = A + (size_t)(m0 + row) * C_DIM;
    const __bf16* aRow1 = A + (size_t)(m0 + 16 + row) * C_DIM;

    v8f acc[2][4] = {{v8f{}, v8f{}, v8f{}, v8f{}},
                     {v8f{}, v8f{}, v8f{}, v8f{}}};

    for (int k0 = 0; k0 < C_DIM; k0 += 32) {
        if (k0 + 32 < C_DIM) {
            __builtin_prefetch(aRow0 + k0 + 32, 0, 3); // WGP-scope global_prefetch_b8
            __builtin_prefetch(aRow1 + k0 + 32, 0, 3);
        }

        // A fragments: lane<16 holds K {0..7,16..23}, lane>=16 holds {8..15,24..31}
        v8bf a0lo = *(const v8bf*)(aRow0 + k0 + sel);
        v8bf a0hi = *(const v8bf*)(aRow0 + k0 + 16 + sel);
        v16bf afrag0 = __builtin_shufflevector(a0lo, a0hi,
            0,1,2,3,4,5,6,7,8,9,10,11,12,13,14,15);
        v8bf a1lo = *(const v8bf*)(aRow1 + k0 + sel);
        v8bf a1hi = *(const v8bf*)(aRow1 + k0 + 16 + sel);
        v16bf afrag1 = __builtin_shufflevector(a1lo, a1hi,
            0,1,2,3,4,5,6,7,8,9,10,11,12,13,14,15);

        #pragma unroll
        for (int t = 0; t < 4; ++t) {
            // B fragment: column n = W row (t*16+row) of the LDS tile,
            // same K selection pattern (B[k,n] = W[n,k]).
            const __bf16* bRow = sW + (size_t)(t * 16 + row) * C_DIM;
            v8bf blo = *(const v8bf*)(bRow + k0 + sel);
            v8bf bhi = *(const v8bf*)(bRow + k0 + 16 + sel);
            v16bf bfrag = __builtin_shufflevector(blo, bhi,
                0,1,2,3,4,5,6,7,8,9,10,11,12,13,14,15);

            acc[0][t] = __builtin_amdgcn_wmma_f32_16x16x32_bf16(
                false, afrag0, false, bfrag, (short)0, acc[0][t], false, false);
            acc[1][t] = __builtin_amdgcn_wmma_f32_16x16x32_bf16(
                false, afrag1, false, bfrag, (short)0, acc[1][t], false, false);
        }
    }

    // C/D layout: VGPR v, lane -> D[M = v + 8*hi][N = lane&15]
    #pragma unroll
    for (int u = 0; u < 2; ++u) {
        #pragma unroll
        for (int t = 0; t < 4; ++t) {
            const int col = n0 + t * 16 + row;
            #pragma unroll
            for (int v = 0; v < 8; ++v) {
                const int m = m0 + u * 16 + v + hi * 8;
                float val = acc[u][t][v];
                if (fuse) {
                    int b = m / N_EDGE;
                    int r = m - b * N_EDGE;
                    int i = r / N_CLS;
                    int j = r - i * N_CLS;
                    val += fA[((size_t)(b * N_CLS + i) << 9) + col]
                         + fB[((size_t)(b * N_CLS + j) << 9) + col];
                }
                out[((size_t)m << 9) + col] = val;
            }
        }
    }
}

// ---------------------------------------------------------------------------
// BatchNorm stats: tensor [B, M, 512]; per-channel m (dim 1) sum & sumsq over
// (batch, C) -> stats[0..M) = sum, stats[M..2M) = sumsq (float atomics).
// ---------------------------------------------------------------------------
__global__ void stats_zero(float* __restrict__ stats, int n) {
    int i = blockIdx.x * blockDim.x + threadIdx.x;
    if (i < n) stats[i] = 0.f;
}

__global__ __launch_bounds__(256)
void bn_stats(const float* __restrict__ t, int B, int M, int bPerBlock,
              float* __restrict__ stats) {
    const int m   = blockIdx.y;
    const int tid = threadIdx.x;
    const int b0  = blockIdx.x * bPerBlock;
    const int b1  = min(B, b0 + bPerBlock);
    float s = 0.f, q = 0.f;
    for (int b = b0; b < b1; ++b) {
        const float* p = t + ((size_t)b * M + m) * C_DIM;
        for (int c = tid; c < C_DIM; c += 256) {
            float v = p[c]; s += v; q += v * v;
        }
    }
    __shared__ float rs[256], rq[256];
    rs[tid] = s; rq[tid] = q;
    __syncthreads();
    for (int o = 128; o > 0; o >>= 1) {
        if (tid < o) { rs[tid] += rs[tid + o]; rq[tid] += rq[tid + o]; }
        __syncthreads();
    }
    if (tid == 0) {
        atomicAdd(&stats[m],     rs[0]);
        atomicAdd(&stats[M + m], rq[0]);
    }
}

__global__ void bn_finalize(float* __restrict__ stats, int M, float invCnt) {
    int m = threadIdx.x;
    if (m < M) {
        float mean = stats[m] * invCnt;
        float var  = stats[M + m] * invCnt - mean * mean;
        stats[m]     = mean;
        stats[M + m] = rsqrtf(var + BN_EPS);
    }
}

// ---------------------------------------------------------------------------
// edge_out = edge_in + relu(bn(agg)); optionally emit bf16 copy for next GEMM.
// ---------------------------------------------------------------------------
__global__ void edge_update(const float4* __restrict__ ein,
                            const float4* __restrict__ agg,
                            const float* __restrict__ stats, int M,
                            float4* __restrict__ eout,
                            ushort4* __restrict__ ebf, size_t n4) {
    size_t i = (size_t)blockIdx.x * blockDim.x + threadIdx.x;
    size_t stride = (size_t)gridDim.x * blockDim.x;
    for (; i < n4; i += stride) {
        size_t f = i << 2;                       // flat element index
        int m = (int)((f >> 9) % (size_t)M);
        float mean = stats[m], rstd = stats[M + m];
        float4 a = agg[i], e = ein[i], o;
        o.x = e.x + fmaxf((a.x - mean) * rstd, 0.f);
        o.y = e.y + fmaxf((a.y - mean) * rstd, 0.f);
        o.z = e.z + fmaxf((a.z - mean) * rstd, 0.f);
        o.w = e.w + fmaxf((a.w - mean) * rstd, 0.f);
        eout[i] = o;
        if (ebf) {
            ushort4 h;
            h.x = f2bf(o.x); h.y = f2bf(o.y); h.z = f2bf(o.z); h.w = f2bf(o.w);
            ebf[i] = h;
        }
    }
}

// ---------------------------------------------------------------------------
// Per (b,i,c): p = softmax_j(sigmoid(edge[b,i*12+j,c]));
// xnew[b,i,c] = Uix[b,i,c] + (sum_j p_j * Vx[b,j,c]) / 12
// ---------------------------------------------------------------------------
__global__ void softmax_msg(const float* __restrict__ edge,
                            const float* __restrict__ Vx,
                            const float* __restrict__ Uix,
                            float* __restrict__ xnew, size_t total) {
    size_t idx = (size_t)blockIdx.x * blockDim.x + threadIdx.x;
    size_t stride = (size_t)gridDim.x * blockDim.x;
    for (; idx < total; idx += stride) {
        const int    c  = (int)(idx & 511);
        const size_t bi = idx >> 9;              // b*12 + i
        const size_t b  = bi / N_CLS;
        const int    i  = (int)(bi - b * N_CLS);
        const float* ep = edge + (((b * N_EDGE) + (size_t)i * N_CLS) << 9) + c;
        float v[N_CLS];
        float mx = -1e30f;
        #pragma unroll
        for (int j = 0; j < N_CLS; ++j) {
            float s = 1.f / (1.f + __expf(-ep[(size_t)j << 9]));
            v[j] = s; mx = fmaxf(mx, s);
        }
        float den = 0.f;
        #pragma unroll
        for (int j = 0; j < N_CLS; ++j) { v[j] = __expf(v[j] - mx); den += v[j]; }
        const float* vp = Vx + ((b * N_CLS) << 9) + c;
        float msg = 0.f;
        #pragma unroll
        for (int j = 0; j < N_CLS; ++j) msg += v[j] * vp[(size_t)j << 9];
        msg /= den;
        xnew[idx] = Uix[idx] + msg * (1.f / (float)N_CLS);
    }
}

// ---------------------------------------------------------------------------
// x_out = relu(res + bn(xnew)); optionally emit bf16 copy for next layer.
// ---------------------------------------------------------------------------
__global__ void x_update(const float4* __restrict__ res,
                         const float4* __restrict__ xnew,
                         const float* __restrict__ stats, int M,
                         float4* __restrict__ xout,
                         ushort4* __restrict__ xbf, size_t n4) {
    size_t i = (size_t)blockIdx.x * blockDim.x + threadIdx.x;
    size_t stride = (size_t)gridDim.x * blockDim.x;
    for (; i < n4; i += stride) {
        size_t f = i << 2;
        int m = (int)((f >> 9) % (size_t)M);
        float mean = stats[m], rstd = stats[M + m];
        float4 a = xnew[i], r = res[i], o;
        o.x = fmaxf(r.x + (a.x - mean) * rstd, 0.f);
        o.y = fmaxf(r.y + (a.y - mean) * rstd, 0.f);
        o.z = fmaxf(r.z + (a.z - mean) * rstd, 0.f);
        o.w = fmaxf(r.w + (a.w - mean) * rstd, 0.f);
        xout[i] = o;
        if (xbf) {
            ushort4 h;
            h.x = f2bf(o.x); h.y = f2bf(o.y); h.z = f2bf(o.z); h.w = f2bf(o.w);
            xbf[i] = h;
        }
    }
}

// ---------------------------------------------------------------------------
// Host orchestration
// ---------------------------------------------------------------------------
extern "C" void kernel_launch(void* const* d_in, const int* in_sizes, int n_in,
                              void* d_out, int out_size, void* d_ws, size_t ws_size,
                              hipStream_t stream) {
    (void)in_sizes; (void)n_in; (void)out_size; (void)ws_size;

    const size_t X = (size_t)X_ROWS * C_DIM;   // 12,582,912
    const size_t E = (size_t)E_ROWS * C_DIM;   // 150,994,944
    const size_t WN = (size_t)C_DIM * C_DIM;   // 262,144

    const float* x_in    = (const float*)d_in[0];
    const float* edge_in = (const float*)d_in[1];
    // d_in[2..11] = WU1, WV1, WA1, WB1, WE1, WU2, WV2, WA2, WB2, WE2

    float* x_out    = (float*)d_out;           // [X]
    float* edge_cur = (float*)d_out + X;       // [E] running/final edge buffer

    // Workspace carve-up
    unsigned char* w = (unsigned char*)d_ws;
    size_t off = 0;
    auto take = [&](size_t bytes) -> void* {
        void* p = w + off;
        off = (off + bytes + 255) & ~(size_t)255;
        return p;
    };
    __bf16* edge_bf = (__bf16*)take(E * 2);
    __bf16* x_bf    = (__bf16*)take(X * 2);
    __bf16* wbf     = (__bf16*)take(10 * WN * 2);
    float*  agg     = (float*)take(E * 4);
    float*  Vix     = (float*)take(X * 4);
    float*  Vjx     = (float*)take(X * 4);
    float*  Uix     = (float*)take(X * 4);
    float*  Vx      = (float*)take(X * 4);
    float*  xnew    = (float*)take(X * 4);
    float*  xcur    = (float*)take(X * 4);
    float*  stats   = (float*)take(1024 * 4);

    const float invCnt = 1.f / ((float)BATCH * (float)C_DIM);  // samples/channel

    auto blocks4 = [](size_t n4) -> unsigned {
        return (unsigned)((n4 + 255) / 256);
    };

    // --- converts -----------------------------------------------------------
    cvt_f32_bf16_v4<<<blocks4(X / 4), 256, 0, stream>>>(
        (const float4*)x_in, (ushort4*)x_bf, X / 4);
    cvt_f32_bf16_v4<<<blocks4(E / 4), 256, 0, stream>>>(
        (const float4*)edge_in, (ushort4*)edge_bf, E / 4);
    for (int wi = 0; wi < 10; ++wi) {
        cvt_f32_bf16_v4<<<blocks4(WN / 4), 256, 0, stream>>>(
            (const float4*)d_in[2 + wi], (ushort4*)(wbf + (size_t)wi * WN), WN / 4);
    }

    auto gemm = [&](const __bf16* A, int wslot, float* out, int Mrows,
                    const float* fa, const float* fb, int fuse) {
        dim3 grid(Mrows / 128, C_DIM / 64);
        gemm_bf16_wmma<<<grid, 128, 0, stream>>>(
            A, wbf + (size_t)wslot * WN, out, Mrows, fa, fb, fuse);
    };

    auto layer = [&](const float* x_res, const float* edge_src,
                     float* edge_dst, __bf16* edge_bf_out,
                     float* x_dst, __bf16* x_bf_out, int wbase) {
        // wbase slots: +0 WU, +1 WV, +2 WA, +3 WB, +4 WE
        gemm(x_bf, wbase + 2, Vix, X_ROWS, nullptr, nullptr, 0);   // WA
        gemm(x_bf, wbase + 3, Vjx, X_ROWS, nullptr, nullptr, 0);   // WB
        gemm(x_bf, wbase + 0, Uix, X_ROWS, nullptr, nullptr, 0);   // WU
        gemm(x_bf, wbase + 1, Vx,  X_ROWS, nullptr, nullptr, 0);   // WV
        gemm(edge_bf, wbase + 4, agg, E_ROWS, Vix, Vjx, 1);        // WE + bcast add

        stats_zero<<<1, 512, 0, stream>>>(stats, 2 * N_EDGE);
        bn_stats<<<dim3(BATCH / 16, N_EDGE), 256, 0, stream>>>(
            agg, BATCH, N_EDGE, 16, stats);
        bn_finalize<<<1, 256, 0, stream>>>(stats, N_EDGE, invCnt);

        edge_update<<<blocks4(E / 4), 256, 0, stream>>>(
            (const float4*)edge_src, (const float4*)agg, stats, N_EDGE,
            (float4*)edge_dst, (ushort4*)edge_bf_out, E / 4);

        softmax_msg<<<blocks4(X), 256, 0, stream>>>(edge_dst, Vx, Uix, xnew, X);

        stats_zero<<<1, 512, 0, stream>>>(stats, 2 * N_CLS);
        bn_stats<<<dim3(BATCH / 16, N_CLS), 256, 0, stream>>>(
            xnew, BATCH, N_CLS, 16, stats);
        bn_finalize<<<1, 256, 0, stream>>>(stats, N_CLS, invCnt);

        x_update<<<blocks4(X / 4), 256, 0, stream>>>(
            (const float4*)x_res, (const float4*)xnew, stats, N_CLS,
            (float4*)x_dst, (ushort4*)x_bf_out, X / 4);
    };

    // Layer 1: res = input x; edge -> edge_cur (d_out region); x -> xcur (ws).
    layer(x_in, edge_in, edge_cur, edge_bf, xcur, x_bf, 0);
    // Layer 2: res = xcur; edge updates in place; x -> d_out. No bf16 needed.
    layer(xcur, edge_cur, edge_cur, nullptr, x_out, nullptr, 5);
}
